// MultiHeadAttention_48404281426029
// MI455X (gfx1250) — matrix operand order repaired and checked by reading, not compiled, not run on hardware
//
#include <hip/hip_runtime.h>

// ---------------------------------------------------------------------------
// MI455X (gfx1250) fused multi-head attention, bf16 WMMA + fp32 accumulate.
// Pipeline: cvt_x -> weight prep -> QKV GEMM -> flash attention -> out proj.
// All operand layouts arranged so every WMMA A/B load is a contiguous b128.
// ---------------------------------------------------------------------------

constexpr int Bb = 8;
constexpr int Tt = 2048;
constexpr int Cc = 384;
constexpr int Hh = 6;
constexpr int Dd = 64;

typedef __attribute__((ext_vector_type(16))) __bf16          v16bf;
typedef __attribute__((ext_vector_type(2)))  __bf16          v2bf;
typedef __attribute__((ext_vector_type(8)))  float           v8f;
typedef __attribute__((ext_vector_type(16))) unsigned short  v16u;
typedef __attribute__((ext_vector_type(8)))  unsigned short  v8u;
typedef __attribute__((ext_vector_type(4)))  unsigned int    v4u32;
typedef __attribute__((ext_vector_type(4)))  float           vf4;

#define DEVI __device__ __forceinline__

// --- packed f32 -> bf16: two floats -> one dword of two bf16 ---------------
DEVI unsigned int f2bf2(float a, float b) {
#if __has_builtin(__builtin_amdgcn_cvt_pk_bf16_f32)
  v2bf t = __builtin_amdgcn_cvt_pk_bf16_f32(a, b);
  return __builtin_bit_cast(unsigned int, t);
#else
  unsigned int r;
  asm("v_cvt_pk_bf16_f32 %0, %1, %2" : "=v"(r) : "v"(a), "v"(b));
  return r;
#endif
}

DEVI unsigned short f2bf(float f) { return (unsigned short)f2bf2(f, 0.f); }

DEVI v8u ld8(const unsigned short* p) { return *(const v8u*)p; }

DEVI v16bf mk16(v8u lo, v8u hi) {
  v16u t;
#pragma unroll
  for (int i = 0; i < 8; ++i) { t[i] = lo[i]; t[8 + i] = hi[i]; }
  return __builtin_bit_cast(v16bf, t);
}

DEVI v8f wmma_bf16(v16bf a, v16bf b, v8f c) {
  // D = A(16x32 bf16) * B(32x16 bf16) + C(16x16 f32)
  return __builtin_amdgcn_wmma_f32_16x16x32_bf16(
      /*neg_a=*/false, a, /*neg_b=*/false, b,
      /*c_mod=*/(short)0, c, /*reuse_a=*/false, /*reuse_b=*/false);
}

DEVI float redmax16(float v) {
#pragma unroll
  for (int m = 1; m <= 8; m <<= 1) v = fmaxf(v, __shfl_xor(v, m, 32));
  return v;  // max over the lane's 16-lane half
}
DEVI float redsum16(float v) {
#pragma unroll
  for (int m = 1; m <= 8; m <<= 1) v += __shfl_xor(v, m, 32);
  return v;
}

// ---------------------------------------------------------------------------
// Kernel C: x f32 -> bf16, 8 elements/thread, fully packed loads/stores
// ---------------------------------------------------------------------------
__global__ __launch_bounds__(256) void k_cvt_x(const float* __restrict__ src,
                                               unsigned short* __restrict__ dst,
                                               long n8) {
  long i = (long)blockIdx.x * blockDim.x + threadIdx.x;
  if (i >= n8) return;
  const float* p = src + i * 8;
  vf4 a = *(const vf4*)p;
  vf4 b = *(const vf4*)(p + 4);
  v4u32 o;
  o[0] = f2bf2(a[0], a[1]);
  o[1] = f2bf2(a[2], a[3]);
  o[2] = f2bf2(b[0], b[1]);
  o[3] = f2bf2(b[2], b[3]);
  *(v4u32*)(dst + i * 8) = o;
}

// ---------------------------------------------------------------------------
// Kernel 0: weight transpose + f32->bf16.  dst[b][c][r] = bf16(src[b][r][c])
// ---------------------------------------------------------------------------
__global__ void k_prep(const float* __restrict__ src, unsigned short* __restrict__ dst,
                       int batch, int rows, int cols) {
  long i = (long)blockIdx.x * blockDim.x + threadIdx.x;
  long total = (long)batch * rows * cols;
  if (i >= total) return;
  int c = (int)(i % cols);
  long t = i / cols;
  int r = (int)(t % rows);
  int b = (int)(t / rows);
  dst[((long)b * cols + c) * rows + r] = f2bf(src[((long)b * rows + r) * cols + c]);
}

// ---------------------------------------------------------------------------
// Kernel 1: QKV projection.  One wave per 16x16 output tile, A/B all bf16.
//   q,k  stored bf16 [b][h][t][d]
//   v    stored bf16 transposed [b][h][d][t]  (PV B-operand friendly)
// ---------------------------------------------------------------------------
__global__ __launch_bounds__(256) void k_qkv(
    const unsigned short* __restrict__ xb,
    const unsigned short* __restrict__ WqT, const unsigned short* __restrict__ WkT,
    const unsigned short* __restrict__ WvT,
    const float* __restrict__ bq, const float* __restrict__ bk, const float* __restrict__ bv,
    unsigned short* __restrict__ q, unsigned short* __restrict__ k,
    unsigned short* __restrict__ vT) {
  const int lane = threadIdx.x & 31;
  const int wv   = threadIdx.x >> 5;
  const int hf   = lane >> 4;
  const int l16  = lane & 15;

  int idx = blockIdx.x * 8 + wv;
  const int sel = idx % 3;  idx /= 3;           // 0=q 1=k 2=v
  const int dt  = idx % (Dd / 16); idx /= (Dd / 16);
  const int tt  = idx % (Tt / 16); idx /= (Tt / 16);
  const int h   = idx % Hh;
  const int b   = idx / Hh;
  if (b >= Bb) return;

  const int t0 = tt * 16, d0 = dt * 16;
  const unsigned short* W = (sel == 0 ? WqT : sel == 1 ? WkT : WvT) + (long)h * Dd * Cc;
  const float* bias       = (sel == 0 ? bq  : sel == 1 ? bk  : bv)  + h * Dd;

  const unsigned short* xr = xb + ((long)b * Tt + t0 + l16) * Cc;  // A row (this lane)
  const unsigned short* wr = W + (long)(d0 + l16) * Cc;            // B col (this lane)

  v8f acc;
#pragma unroll
  for (int i = 0; i < 8; ++i) acc[i] = 0.f;

#pragma unroll 4
  for (int k0 = 0; k0 < Cc; k0 += 32) {
    v16bf A  = mk16(ld8(xr + k0 + hf * 8), ld8(xr + k0 + 16 + hf * 8));
    v16bf Bm = mk16(ld8(wr + k0 + hf * 8), ld8(wr + k0 + 16 + hf * 8));
    acc = wmma_bf16(A, Bm, acc);
  }

  const float bval = bias[d0 + l16];

  if (sel == 2) {
    // transposed store: vT[b][h][d][t], 8 consecutive t per lane -> one b128
    const int d = d0 + l16;
    unsigned short* dst = vT + (((long)b * Hh + h) * Dd + d) * Tt + t0 + 8 * hf;
    v4u32 o;
#pragma unroll
    for (int r = 0; r < 4; ++r)
      o[r] = f2bf2(acc[2 * r] + bval, acc[2 * r + 1] + bval);
    *(v4u32*)dst = o;
  } else {
    unsigned short* base = (sel == 0 ? q : k) + ((long)b * Hh + h) * Tt * Dd
                           + (long)t0 * Dd + d0 + l16;
#pragma unroll
    for (int r = 0; r < 8; r += 2) {
      const unsigned int pk = f2bf2(acc[r] + bval, acc[r + 1] + bval);
      base[(long)(r + 8 * hf) * Dd]     = (unsigned short)pk;          // store b16
      base[(long)(r + 1 + 8 * hf) * Dd] = (unsigned short)(pk >> 16);  // d16_hi store
    }
  }
}

// ---------------------------------------------------------------------------
// Kernel 2: causal flash attention.  One wave per 16-row query tile.
//   Online softmax stats mi/li indexed by r match WMMA C-layout rows exactly:
//   row = r + 8*(lane>=16).
// ---------------------------------------------------------------------------
__global__ __launch_bounds__(256) void k_attn(
    const unsigned short* __restrict__ q, const unsigned short* __restrict__ k,
    const unsigned short* __restrict__ vT, unsigned short* __restrict__ ao) {
  __shared__ __align__(16) unsigned short pbuf[8][16][32];   // per-wave P transpose buffer

  const int lane = threadIdx.x & 31;
  const int wv   = threadIdx.x >> 5;
  const int hf   = lane >> 4;
  const int l16  = lane & 15;

  int idx = blockIdx.x * 8 + wv;
  const int tt = idx % (Tt / 16); idx /= (Tt / 16);
  const int h  = idx % Hh;
  const int b  = idx / Hh;
  if (b >= Bb) return;

  const int  t0 = tt * 16;
  const long bh = (long)b * Hh + h;
  const unsigned short* qp = q  + bh * Tt * Dd;
  const unsigned short* kp = k  + bh * Tt * Dd;
  const unsigned short* vp = vT + bh * Dd * Tt;

  // Q A-operand tiles (16 t-rows x 64 d), loaded once
  v16bf aq[2];
  {
    const unsigned short* row = qp + (long)(t0 + l16) * Dd;
#pragma unroll
    for (int dk = 0; dk < 2; ++dk)
      aq[dk] = mk16(ld8(row + dk * 32 + hf * 8), ld8(row + dk * 32 + 16 + hf * 8));
  }

  float mi[8], li[8];
  v8f acc[4];
#pragma unroll
  for (int r = 0; r < 8; ++r) { mi[r] = -1e30f; li[r] = 0.f; }
#pragma unroll
  for (int d = 0; d < 4; ++d)
#pragma unroll
    for (int r = 0; r < 8; ++r) acc[d][r] = 0.f;

  const float sc = 0.125f * 1.44269504088896f;   // D^-0.5 * log2(e)
  const int nsb = tt / 2 + 1;                    // causal: key blocks of 32

  for (int sb = 0; sb < nsb; ++sb) {
    const int s0 = sb * 32;
    const bool diag = (sb == nsb - 1);

    if (sb + 1 < nsb) {   // prefetch next K block -> global_prefetch_b8
      __builtin_prefetch(kp + (long)(s0 + 32 + l16) * Dd, 0, 1);
      __builtin_prefetch(vp + (long)l16 * Tt + s0 + 32, 0, 1);
    }

    // S = Q K^T for two 16-key subtiles (contraction over d = 2 WMMAs each)
    v8f S[2];
#pragma unroll
    for (int st = 0; st < 2; ++st) {
      v8f s;
#pragma unroll
      for (int i = 0; i < 8; ++i) s[i] = 0.f;
      const unsigned short* kr = kp + (long)(s0 + st * 16 + l16) * Dd;
#pragma unroll
      for (int dk = 0; dk < 2; ++dk)
        s = wmma_bf16(aq[dk], mk16(ld8(kr + dk * 32 + hf * 8),
                                   ld8(kr + dk * 32 + 16 + hf * 8)), s);
      S[st] = s;
    }

    if (diag) {
#pragma unroll
      for (int st = 0; st < 2; ++st)
#pragma unroll
        for (int r = 0; r < 8; ++r) {
          const int row = t0 + r + 8 * hf;
          const int col = s0 + st * 16 + l16;
          if (col > row) S[st][r] = -1e30f;
        }
    }

    // online softmax update (per C-layout row r)
    float esc[8];
#pragma unroll
    for (int r = 0; r < 8; ++r) {
      float mx = redmax16(fmaxf(S[0][r], S[1][r]));
      const float mnew = fmaxf(mi[r], mx);
      esc[r] = __builtin_amdgcn_exp2f((mi[r] - mnew) * sc);
      mi[r] = mnew;
      const float p0 = __builtin_amdgcn_exp2f((S[0][r] - mnew) * sc);
      const float p1 = __builtin_amdgcn_exp2f((S[1][r] - mnew) * sc);
      S[0][r] = p0; S[1][r] = p1;
      li[r] = li[r] * esc[r] + redsum16(p0 + p1);
    }
#pragma unroll
    for (int d = 0; d < 4; ++d)
#pragma unroll
      for (int r = 0; r < 8; ++r) acc[d][r] *= esc[r];

    // P: C-layout -> A-layout via LDS (wave-private region; LDS is in-order).
    // Packed cvt; hi half stored via d16_hi.
#pragma unroll
    for (int r = 0; r < 8; ++r) {
      const unsigned int pk = f2bf2(S[0][r], S[1][r]);
      unsigned short* pr = &pbuf[wv][r + 8 * hf][l16];
      pr[0]  = (unsigned short)pk;
      pr[16] = (unsigned short)(pk >> 16);
    }
    __builtin_amdgcn_wave_barrier();
    asm volatile("s_wait_dscnt 0" ::: "memory");

    v16bf ap;
    {
      const unsigned short* pr = &pbuf[wv][l16][0];
      ap = mk16(*(const v8u*)(pr + hf * 8), *(const v8u*)(pr + 16 + hf * 8));
    }
    __builtin_amdgcn_wave_barrier();

    // O += P V   (4 d-tiles; V read from transposed layout, contiguous b128s)
#pragma unroll
    for (int d = 0; d < 4; ++d) {
      const unsigned short* vr = vp + (long)(d * 16 + l16) * Tt + s0 + hf * 8;
      acc[d] = wmma_bf16(ap, mk16(ld8(vr), ld8(vr + 16)), acc[d]);
    }
  }

  // normalize + store attn output bf16 [b][t][h*64+d]
#pragma unroll
  for (int r = 0; r < 8; ++r) li[r] = 1.0f / li[r];
#pragma unroll
  for (int dp = 0; dp < 2; ++dp)
#pragma unroll
    for (int r = 0; r < 8; ++r) {
      const unsigned int pk = f2bf2(acc[2 * dp][r] * li[r], acc[2 * dp + 1][r] * li[r]);
      unsigned short* dst = ao + ((long)b * Tt + t0 + r + 8 * hf) * Cc
                               + h * Dd + dp * 32 + l16;
      dst[0]  = (unsigned short)pk;
      dst[16] = (unsigned short)(pk >> 16);
    }
}

// ---------------------------------------------------------------------------
// Kernel 3: output projection  out = ao @ Wp + bp   (fp32 out)
// ---------------------------------------------------------------------------
__global__ __launch_bounds__(256) void k_proj(
    const unsigned short* __restrict__ ao, const unsigned short* __restrict__ WpT,
    const float* __restrict__ bp, float* __restrict__ out) {
  const int lane = threadIdx.x & 31;
  const int wv   = threadIdx.x >> 5;
  const int hf   = lane >> 4;
  const int l16  = lane & 15;

  int idx = blockIdx.x * 8 + wv;
  const int ct = idx % (Cc / 16);
  const int mt = idx / (Cc / 16);
  if (mt >= (Bb * Tt) / 16) return;
  const int r0 = mt * 16, c0 = ct * 16;

  const unsigned short* ar = ao  + ((long)r0 + l16) * Cc;
  const unsigned short* wr = WpT + (long)(c0 + l16) * Cc;

  v8f acc;
#pragma unroll
  for (int i = 0; i < 8; ++i) acc[i] = 0.f;

#pragma unroll 4
  for (int k0 = 0; k0 < Cc; k0 += 32) {
    v16bf A  = mk16(ld8(ar + k0 + hf * 8), ld8(ar + k0 + 16 + hf * 8));
    v16bf Bm = mk16(ld8(wr + k0 + hf * 8), ld8(wr + k0 + 16 + hf * 8));
    acc = wmma_bf16(A, Bm, acc);
  }

  const float bval = bp[c0 + l16];
#pragma unroll
  for (int r = 0; r < 8; ++r)
    out[((long)r0 + r + 8 * hf) * Cc + c0 + l16] = acc[r] + bval;
}

// ---------------------------------------------------------------------------
// Host launcher
// ---------------------------------------------------------------------------
extern "C" void kernel_launch(void* const* d_in, const int* in_sizes, int n_in,
                              void* d_out, int out_size, void* d_ws, size_t ws_size,
                              hipStream_t stream) {
  const float* x  = (const float*)d_in[0];
  const float* Wq = (const float*)d_in[1];
  const float* bq = (const float*)d_in[2];
  const float* Wk = (const float*)d_in[3];
  const float* bk = (const float*)d_in[4];
  const float* Wv = (const float*)d_in[5];
  const float* bv = (const float*)d_in[6];
  const float* Wp = (const float*)d_in[7];
  const float* bp = (const float*)d_in[8];
  (void)in_sizes; (void)n_in; (void)out_size; (void)ws_size;

  // workspace layout (bf16 ushorts):  xb | q | k | vT | ao | WqT | WkT | WvT | WpT
  const size_t QS = (size_t)Bb * Hh * Tt * Dd;   // 6,291,456  ( == B*T*C )
  const size_t WS = (size_t)Hh * Cc * Dd;        // 147,456    ( == C*C )
  unsigned short* wsb = (unsigned short*)d_ws;
  unsigned short* xbb = wsb;
  unsigned short* qb  = wsb + 1 * QS;
  unsigned short* kb  = wsb + 2 * QS;
  unsigned short* vTb = wsb + 3 * QS;
  unsigned short* aob = wsb + 4 * QS;
  unsigned short* WqT = wsb + 5 * QS;
  unsigned short* WkT = WqT + WS;
  unsigned short* WvT = WkT + WS;
  unsigned short* WpT = WvT + WS;

  // C) x f32 -> bf16 (one-time; removes conversions from GEMM hot loops)
  {
    const long n8 = (long)Bb * Tt * Cc / 8;      // 786,432
    k_cvt_x<<<(int)((n8 + 255) / 256), 256, 0, stream>>>(x, xbb, n8);
  }

  // 0) weight prep: transpose + convert to bf16
  {
    const int tot = Hh * Cc * Dd;
    const int g = (tot + 255) / 256;
    k_prep<<<g, 256, 0, stream>>>(Wq, WqT, Hh, Cc, Dd);
    k_prep<<<g, 256, 0, stream>>>(Wk, WkT, Hh, Cc, Dd);
    k_prep<<<g, 256, 0, stream>>>(Wv, WvT, Hh, Cc, Dd);
    const int totp = Cc * Cc;
    k_prep<<<(totp + 255) / 256, 256, 0, stream>>>(Wp, WpT, 1, Cc, Cc);
  }

  // 1) QKV projection: B*H*(T/16)*(D/16)*3 tiles, 8 waves/block
  {
    const int tiles = Bb * Hh * (Tt / 16) * (Dd / 16) * 3;   // 73728
    k_qkv<<<tiles / 8, 256, 0, stream>>>(xbb, WqT, WkT, WvT, bq, bk, bv, qb, kb, vTb);
  }

  // 2) flash attention: B*H*(T/16) query tiles, 8 waves/block
  {
    const int tiles = Bb * Hh * (Tt / 16);                   // 6144
    k_attn<<<tiles / 8, 256, 0, stream>>>(qb, kb, vTb, aob);
  }

  // 3) output projection: (B*T/16)*(C/16) tiles
  {
    const int tiles = (Bb * Tt / 16) * (Cc / 16);            // 24576
    k_proj<<<tiles / 8, 256, 0, stream>>>(aob, WpT, bp, (float*)d_out);
  }
}
